// ExtendSpGAT_1400159339119
// MI455X (gfx1250) — compile-verified, matrix operand order checked by reading, not run on hardware
//
#include <hip/hip_runtime.h>

#define NNODE 50000
#define NEDGE 800000
#define NFEATD 256
#define NHID 64
#define NHEADS 8
#define NCLASS 40
#define HCATW (NHEADS * NHID) /* 512 */
#define HA 0.1f
#define LRELU 0.2f
#define EPSV 1e-16f

typedef __attribute__((ext_vector_type(16))) __bf16 v16bf;
typedef __attribute__((ext_vector_type(8)))  float  v8f;
typedef __attribute__((ext_vector_type(8)))  unsigned int v8u;
typedef __attribute__((ext_vector_type(4)))  unsigned int u32x4;
typedef __attribute__((ext_vector_type(8)))  int i32x8;
typedef __attribute__((ext_vector_type(4)))  int i32x4;

__device__ __forceinline__ unsigned short f2bf_u16(float f) {
  unsigned u = __builtin_bit_cast(unsigned, f);
  u += 0x7FFFu + ((u >> 16) & 1u); // RNE to bf16
  return (unsigned short)(u >> 16);
}
__device__ __forceinline__ __bf16 f2bf(float f) {
  unsigned short h = f2bf_u16(f);
  return __builtin_bit_cast(__bf16, h);
}
__device__ __forceinline__ float eluf(float x) {
  return x > 0.f ? x : (__expf(x) - 1.0f);
}

// ---------------------------------------------------------------------------
// Pack B into WMMA bf16 fragment order: frag[ct][kc][lane][j] so the GEMM
// loads one aligned 32B vector per (tile, k-chunk, lane). Zero-padded cols.
// koff(j,lane) per CDNA5 ISA 7.12.2 16-bit B 32x16 layout.
// ---------------------------------------------------------------------------
__global__ void pack_bfrag_w1(const float* __restrict__ W1, unsigned short* __restrict__ Bf) {
  const int KC = NFEATD / 32;                 // 8
  int i = blockIdx.x * blockDim.x + threadIdx.x;  // over Ntiles*KC*32
  if (i >= 32 * KC * 32) return;
  int lane = i & 31, kc = (i >> 5) % KC, ct = (i >> 5) / KC;
  int kb = (lane >> 4) << 3;
  int col = ct * 16 + (lane & 15);            // < 512 always
  int h = col >> 6, f = col & 63;
  unsigned short* o = Bf + (size_t)i * 16;
#pragma unroll
  for (int j = 0; j < 16; ++j) {
    int k = kc * 32 + ((j >> 3) << 4) + kb + (j & 7);
    o[j] = f2bf_u16(W1[((size_t)h * NFEATD + k) * NHID + f]);
  }
}
__global__ void pack_bfrag_w2(const float* __restrict__ W2, unsigned short* __restrict__ Bf) {
  const int KC = HCATW / 32;                  // 16
  int i = blockIdx.x * blockDim.x + threadIdx.x;  // over 3*KC*32
  if (i >= 3 * KC * 32) return;
  int lane = i & 31, kc = (i >> 5) % KC, ct = (i >> 5) / KC;
  int kb = (lane >> 4) << 3;
  int col = ct * 16 + (lane & 15);
  unsigned short* o = Bf + (size_t)i * 16;
#pragma unroll
  for (int j = 0; j < 16; ++j) {
    int k = kc * 32 + ((j >> 3) << 4) + kb + (j & 7);
    o[j] = (col < NCLASS) ? f2bf_u16(W2[(size_t)k * NCLASS + col]) : (unsigned short)0;
  }
}

// ---------------------------------------------------------------------------
// bf16-WMMA GEMM with TDM-staged A tile:
//   wave 0 DMAs A[m0:m0+16, 0:K] (f32, stride lda) into LDS via
//   tensor_load_to_lds, waits on TENSORcnt, then 8 waves consume it.
//   Each wave computes TPW 16x16 C tiles (A fragment reused TPW times).
// ---------------------------------------------------------------------------
template <int TPW>
__global__ __launch_bounds__(256) void wmma_gemm_tdm(
    const float* __restrict__ A, const unsigned short* __restrict__ Bf,
    float* __restrict__ C, int M, int K, int Ntiles, int Ncols, int lda, int ldc) {
  __shared__ float shA[16 * 512];             // sole LDS array -> offset 0
  const int wave = threadIdx.x >> 5;
  const int lane = threadIdx.x & 31;
  const int m0 = blockIdx.x << 4;

  if (wave == 0) {
    unsigned long long ga = (unsigned long long)(const void*)(A + (size_t)m0 * lda);
    u32x4 g0;
    g0[0] = 1u;                               // count=1 valid descriptor
    g0[1] = 0u;                               // lds_addr (bytes) = offset of shA
    g0[2] = (unsigned)(ga & 0xFFFFFFFFull);   // global_addr lo
    g0[3] = (unsigned)(ga >> 32) | (2u << 30);// global_addr hi | type=2
    i32x8 g1;
    g1[0] = (2 << 16);                        // data_size=4B; no multicast/pad
    g1[1] = (int)((K & 0xFFFF) << 16);        // tensor_dim0[15:0]  @bits[63:48]
    g1[2] = (int)(((K >> 16) & 0xFFFF) | (16 << 16)); // dim0 hi | tensor_dim1 lo (=16 rows)
    g1[3] = (int)((K & 0xFFFF) << 16);        // tensor_dim1 hi=0 | tile_dim0=K
    g1[4] = 16;                               // tile_dim1=16, tile_dim2=0
    g1[5] = lda;                              // tensor_dim0_stride lo32 (elements)
    g1[6] = 0;                                // stride hi | dim1_stride lo
    g1[7] = 0;
    i32x4 z4 = {0, 0, 0, 0};
    i32x8 z8 = {0, 0, 0, 0, 0, 0, 0, 0};
    __builtin_amdgcn_tensor_load_to_lds(g0, g1, z4, z4, z8, 0);
    __builtin_amdgcn_s_wait_tensorcnt(0);
  }
  __syncthreads();

  const int KC = K >> 5;
  const int ct0 = wave * TPW;
  if (ct0 >= Ntiles) return;                  // after the barrier: safe

  const int kb = (lane >> 4) << 3;            // 0 or 8
  v8f zero = {};
  v8f acc[TPW];
#pragma unroll
  for (int t = 0; t < TPW; ++t) acc[t] = zero;

  const float* arow = shA + (lane & 15) * K;
  for (int kc = 0; kc < KC; ++kc) {
    const float* ar = arow + (kc << 5);
    float4 p0 = *(const float4*)(ar + kb);
    float4 p1 = *(const float4*)(ar + kb + 4);
    float4 p2 = *(const float4*)(ar + kb + 16);
    float4 p3 = *(const float4*)(ar + kb + 20);
    v16bf af;
    af[0] = f2bf(p0.x); af[1] = f2bf(p0.y); af[2]  = f2bf(p0.z); af[3]  = f2bf(p0.w);
    af[4] = f2bf(p1.x); af[5] = f2bf(p1.y); af[6]  = f2bf(p1.z); af[7]  = f2bf(p1.w);
    af[8] = f2bf(p2.x); af[9] = f2bf(p2.y); af[10] = f2bf(p2.z); af[11] = f2bf(p2.w);
    af[12] = f2bf(p3.x); af[13] = f2bf(p3.y); af[14] = f2bf(p3.z); af[15] = f2bf(p3.w);
#pragma unroll
    for (int t = 0; t < TPW; ++t) {
      const v8u bv = *(const v8u*)(Bf + ((((size_t)(ct0 + t) * KC + kc) << 5) + lane) * 16);
      v16bf bfr = __builtin_bit_cast(v16bf, bv);
      acc[t] = __builtin_amdgcn_wmma_f32_16x16x32_bf16(
          false, af, false, bfr, (short)0, acc[t], false, false);
    }
  }

  const int rbase = m0 + ((lane >> 4) << 3);
#pragma unroll
  for (int t = 0; t < TPW; ++t) {
    const int ncol = (ct0 + t) * 16 + (lane & 15);
    if (ncol < Ncols) {
#pragma unroll
      for (int r = 0; r < 8; ++r) {
        const int row = rbase + r;
        if (row < M) C[(size_t)row * ldc + ncol] = acc[t][r];
      }
    }
  }
}

__global__ void fill_f32(float* p, float v, int n) {
  int i = blockIdx.x * blockDim.x + threadIdx.x;
  if (i < n) p[i] = v;
}
__global__ void fill_i32(int* p, int v, int n) {
  int i = blockIdx.x * blockDim.x + threadIdx.x;
  if (i < n) p[i] = v;
}
__global__ void copy_i32(const int* s, int* d, int n) {
  int i = blockIdx.x * blockDim.x + threadIdx.x;
  if (i < n) d[i] = s[i];
}

__global__ void node_alpha8(const float* __restrict__ H, const float* __restrict__ a1,
                            float* __restrict__ asrc, float* __restrict__ adst) {
  int t = blockIdx.x * blockDim.x + threadIdx.x;
  if (t >= NNODE * NHEADS) return;
  int h = t & 7, n = t >> 3;
  const float* hr = H + (size_t)n * HCATW + h * NHID;
  const float* av = a1 + (size_t)h * (2 * NHID);
  float s0 = 0.f, s1 = 0.f;
#pragma unroll 8
  for (int f = 0; f < NHID; ++f) { float v = hr[f]; s0 += v * av[f]; s1 += v * av[NHID + f]; }
  asrc[(size_t)h * NNODE + n] = s0;
  adst[(size_t)h * NNODE + n] = s1;
}

__global__ void edge_e8(const int* __restrict__ src, const int* __restrict__ dst,
                        const float* __restrict__ asrc, const float* __restrict__ adst,
                        float* __restrict__ e1, float* __restrict__ rs) {
  int e = blockIdx.x * blockDim.x + threadIdx.x;
  if (e >= NEDGE) return;
  int s = src[e], d = dst[e];
#pragma unroll
  for (int h = 0; h < NHEADS; ++h) {
    float lg = asrc[(size_t)h * NNODE + s] + adst[(size_t)h * NNODE + d];
    float lr = lg > 0.f ? lg : LRELU * lg;
    float ev = __expf(-lr);
    e1[(size_t)h * NEDGE + e] = ev;
    atomicAdd(&rs[(size_t)h * NNODE + d], ev);
  }
}

__global__ void deg_count(const int* __restrict__ dst, int* __restrict__ deg) {
  int e = blockIdx.x * blockDim.x + threadIdx.x;
  if (e < NEDGE) atomicAdd(&deg[dst[e]], 1);
}

__global__ void scan_block(const int* __restrict__ deg, int* __restrict__ part,
                           int* __restrict__ bsum) {
  __shared__ int sh[256];
  int i = blockIdx.x * 256 + threadIdx.x;
  sh[threadIdx.x] = (i < NNODE) ? deg[i] : 0;
  __syncthreads();
  for (int off = 1; off < 256; off <<= 1) {
    int t = (threadIdx.x >= (unsigned)off) ? sh[threadIdx.x - off] : 0;
    __syncthreads();
    sh[threadIdx.x] += t;
    __syncthreads();
  }
  if (i < NNODE) part[i] = sh[threadIdx.x];
  if (threadIdx.x == 255) bsum[blockIdx.x] = sh[255];
}
__global__ void scan_bsum(int* bsum, int nb) {  // nb <= 256 (196 here)
  __shared__ int sh[256];
  int i = threadIdx.x;
  sh[i] = (i < nb) ? bsum[i] : 0;
  __syncthreads();
  for (int off = 1; off < 256; off <<= 1) {
    int t = (i >= off) ? sh[i - off] : 0;
    __syncthreads();
    sh[i] += t;
    __syncthreads();
  }
  if (i < nb) bsum[i] = sh[i];
}
__global__ void scan_final(const int* __restrict__ part, const int* __restrict__ bsum,
                           int* __restrict__ rowptr) {
  int i = blockIdx.x * 256 + threadIdx.x;
  if (i < NNODE)
    rowptr[i + 1] = part[i] + (blockIdx.x > 0 ? bsum[blockIdx.x - 1] : 0);
  if (i == 0) rowptr[0] = 0;
}

__global__ void csr_fill(const int* __restrict__ src, const int* __restrict__ dst,
                         int* __restrict__ cursor, int* __restrict__ csrc,
                         int* __restrict__ ceid) {
  int e = blockIdx.x * blockDim.x + threadIdx.x;
  if (e >= NEDGE) return;
  int slot = atomicAdd(&cursor[dst[e]], 1);
  csrc[slot] = src[e];
  ceid[slot] = e;
}

__global__ void reorder_e8(const float* __restrict__ e1, const int* __restrict__ ceid,
                           float* __restrict__ e1c) {
  int idx = blockIdx.x * blockDim.x + threadIdx.x;
  if (idx >= NEDGE) return;
  int eid = ceid[idx];
#pragma unroll
  for (int h = 0; h < NHEADS; ++h)
    e1c[(size_t)h * NEDGE + idx] = e1[(size_t)h * NEDGE + eid];
}
__global__ void reorder_e1(const float* __restrict__ e2, const int* __restrict__ ceid,
                           float* __restrict__ e2c) {
  int idx = blockIdx.x * blockDim.x + threadIdx.x;
  if (idx < NEDGE) e2c[idx] = e2[ceid[idx]];
}

// One wave per (node, head): atomic-free CSR gather hop; working set L2-resident.
__global__ __launch_bounds__(256) void hop_l1(
    const float* __restrict__ feat, const float* __restrict__ Hb,
    const float* __restrict__ ec, const float* __restrict__ rs,
    const int* __restrict__ rowptr, const int* __restrict__ csrc,
    float* __restrict__ out, int apply_elu) {
  const int n = blockIdx.x;
  const int h = threadIdx.x >> 5;
  const int lane = threadIdx.x & 31;
  const int st = rowptr[n], en = rowptr[n + 1];
  const float* eh = ec + (size_t)h * NEDGE;
  float a0 = 0.f, a1 = 0.f;
  for (int idx = st; idx < en; ++idx) {
    if (idx + 1 < en)
      __builtin_prefetch(feat + (size_t)csrc[idx + 1] * HCATW + h * NHID, 0, 0);
    const int s = csrc[idx];
    const float ev = eh[idx];
    const float* fr = feat + (size_t)s * HCATW + h * NHID;
    a0 += ev * fr[lane];
    a1 += ev * fr[lane + 32];
  }
  const float inv = 1.0f / rs[(size_t)h * NNODE + n];
  const float* hr = Hb + (size_t)n * HCATW + h * NHID;
  float v0 = (1.0f - HA) * a0 * inv + HA * hr[lane];
  float v1 = (1.0f - HA) * a1 * inv + HA * hr[lane + 32];
  if (apply_elu) { v0 = eluf(v0); v1 = eluf(v1); }
  float* orow = out + (size_t)n * HCATW + h * NHID;
  orow[lane] = v0;
  orow[lane + 32] = v1;
}

__global__ void node_alpha2(const float* __restrict__ h2, const float* __restrict__ a2,
                            float* __restrict__ as2, float* __restrict__ ad2) {
  int n = blockIdx.x * blockDim.x + threadIdx.x;
  if (n >= NNODE) return;
  const float* hr = h2 + (size_t)n * NCLASS;
  float s0 = 0.f, s1 = 0.f;
#pragma unroll 8
  for (int f = 0; f < NCLASS; ++f) { float v = hr[f]; s0 += v * a2[f]; s1 += v * a2[NCLASS + f]; }
  as2[n] = s0;
  ad2[n] = s1;
}

__global__ void edge_e1k(const int* __restrict__ src, const int* __restrict__ dst,
                         const float* __restrict__ as2, const float* __restrict__ ad2,
                         float* __restrict__ e2, float* __restrict__ rs2) {
  int e = blockIdx.x * blockDim.x + threadIdx.x;
  if (e >= NEDGE) return;
  float lg = as2[src[e]] + ad2[dst[e]];
  float lr = lg > 0.f ? lg : LRELU * lg;
  float ev = __expf(-lr);
  e2[e] = ev;
  atomicAdd(&rs2[dst[e]], ev);
}

__global__ __launch_bounds__(256) void hop_l2(
    const float* __restrict__ feat, const float* __restrict__ Hb,
    const float* __restrict__ ec, const float* __restrict__ rs,
    const int* __restrict__ rowptr, const int* __restrict__ csrc,
    float* __restrict__ out) {
  const int n = blockIdx.x * 8 + (threadIdx.x >> 5);
  if (n >= NNODE) return;
  const int lane = threadIdx.x & 31;
  const bool hi = lane < (NCLASS - 32);
  float a0 = 0.f, a1 = 0.f;
  const int st = rowptr[n], en = rowptr[n + 1];
  for (int idx = st; idx < en; ++idx) {
    const int s = csrc[idx];
    const float ev = ec[idx];
    const float* fr = feat + (size_t)s * NCLASS;
    a0 += ev * fr[lane];
    if (hi) a1 += ev * fr[lane + 32];
  }
  const float inv = 1.0f / rs[n];
  const float* hr = Hb + (size_t)n * NCLASS;
  float* orow = out + (size_t)n * NCLASS;
  orow[lane] = (1.0f - HA) * a0 * inv + HA * hr[lane];
  if (hi) orow[lane + 32] = (1.0f - HA) * a1 * inv + HA * hr[lane + 32];
}

__global__ __launch_bounds__(256) void elu_logsoftmax(
    const float* __restrict__ feat, float* __restrict__ out) {
  const int n = blockIdx.x * 8 + (threadIdx.x >> 5);
  if (n >= NNODE) return;
  const int lane = threadIdx.x & 31;
  const bool hi = lane < (NCLASS - 32);
  const float* fr = feat + (size_t)n * NCLASS;
  float x0 = eluf(fr[lane]);
  float x1 = hi ? eluf(fr[lane + 32]) : 0.f;
  float m = hi ? fmaxf(x0, x1) : x0;
#pragma unroll
  for (int off = 16; off; off >>= 1) m = fmaxf(m, __shfl_xor(m, off, 32));
  float s = __expf(x0 - m) + (hi ? __expf(x1 - m) : 0.f);
#pragma unroll
  for (int off = 16; off; off >>= 1) s += __shfl_xor(s, off, 32);
  float lse = m + __logf(s);
  out[(size_t)n * NCLASS + lane] = x0 - lse;
  if (hi) out[(size_t)n * NCLASS + lane + 32] = x1 - lse;
}

extern "C" void kernel_launch(void* const* d_in, const int* in_sizes, int n_in,
                              void* d_out, int out_size, void* d_ws, size_t ws_size,
                              hipStream_t stream) {
  (void)in_sizes; (void)n_in; (void)out_size; (void)ws_size;
  const float* x  = (const float*)d_in[0];
  const int*   ei = (const int*)d_in[1];
  const float* W1 = (const float*)d_in[2];
  const float* a1 = (const float*)d_in[3];
  const float* W2 = (const float*)d_in[4];
  const float* a2 = (const float*)d_in[5];
  const int* src = ei;
  const int* dst = ei + NEDGE;
  float* out = (float*)d_out;

  char* p = (char*)d_ws;
  auto alloc = [&](size_t bytes) -> void* {
    void* r = (void*)p;
    p += (bytes + 255) & ~(size_t)255;
    return r;
  };
  float* H      = (float*)alloc((size_t)NNODE * HCATW * 4);
  float* fA     = (float*)alloc((size_t)NNODE * HCATW * 4);
  float* fB     = (float*)alloc((size_t)NNODE * HCATW * 4);
  unsigned short* Bf1 = (unsigned short*)alloc((size_t)32 * 8 * 32 * 16 * 2);
  unsigned short* Bf2 = (unsigned short*)alloc((size_t)3 * 16 * 32 * 16 * 2);
  float* asrc   = (float*)alloc((size_t)NHEADS * NNODE * 4);
  float* adst   = (float*)alloc((size_t)NHEADS * NNODE * 4);
  float* e1     = (float*)alloc((size_t)NHEADS * NEDGE * 4);
  float* e1c    = (float*)alloc((size_t)NHEADS * NEDGE * 4);
  float* rs1    = (float*)alloc((size_t)NHEADS * NNODE * 4);
  int*   deg    = (int*)alloc((size_t)NNODE * 4);
  int*   bsum   = (int*)alloc(4096);
  int*   rowptr = (int*)alloc((size_t)(NNODE + 1) * 4);
  int*   cursor = (int*)alloc((size_t)NNODE * 4);
  int*   part   = (int*)alloc((size_t)NNODE * 4);
  int*   csrc   = (int*)alloc((size_t)NEDGE * 4);
  int*   ceid   = (int*)alloc((size_t)NEDGE * 4);
  float* h2     = (float*)alloc((size_t)NNODE * NCLASS * 4);
  float* f2A    = (float*)alloc((size_t)NNODE * NCLASS * 4);
  float* f2B    = (float*)alloc((size_t)NNODE * NCLASS * 4);
  float* as2    = (float*)alloc((size_t)NNODE * 4);
  float* ad2    = (float*)alloc((size_t)NNODE * 4);
  float* e2     = (float*)alloc((size_t)NEDGE * 4);
  float* e2c    = (float*)alloc((size_t)NEDGE * 4);
  float* rs2    = (float*)alloc((size_t)NNODE * 4);

  const int TB = 256;
  dim3 b(TB);
  const int gE = (NEDGE + TB - 1) / TB;
  const int gN = (NNODE + TB - 1) / TB;  // 196 blocks -> single-block level-2 scan ok

  // B fragment packing (bf16, padded, WMMA-native layout)
  pack_bfrag_w1<<<(32 * 8 * 32 + TB - 1) / TB, b, 0, stream>>>(W1, Bf1);
  pack_bfrag_w2<<<(3 * 16 * 32 + TB - 1) / TB, b, 0, stream>>>(W2, Bf2);

  // Layer 1 GEMM: H[N x 512] = x[N x 256] @ W1  (TDM A-stage + bf16 WMMA)
  wmma_gemm_tdm<4><<<dim3(NNODE / 16), b, 0, stream>>>(
      x, Bf1, H, NNODE, NFEATD, 32, HCATW, NFEATD, HCATW);

  // Attention coefficients + per-edge e + rowsum
  node_alpha8<<<(NNODE * NHEADS + TB - 1) / TB, b, 0, stream>>>(H, a1, asrc, adst);
  fill_f32<<<(NHEADS * NNODE + TB - 1) / TB, b, 0, stream>>>(rs1, EPSV, NHEADS * NNODE);
  fill_f32<<<gN, b, 0, stream>>>(rs2, EPSV, NNODE);
  fill_i32<<<gN, b, 0, stream>>>(deg, 0, NNODE);
  edge_e8<<<gE, b, 0, stream>>>(src, dst, asrc, adst, e1, rs1);

  // CSR build (dst-sorted) -> atomic-free gather hops
  deg_count<<<gE, b, 0, stream>>>(dst, deg);
  scan_block<<<gN, b, 0, stream>>>(deg, part, bsum);
  scan_bsum<<<1, b, 0, stream>>>(bsum, gN);
  scan_final<<<gN, b, 0, stream>>>(part, bsum, rowptr);
  copy_i32<<<gN, b, 0, stream>>>(rowptr, cursor, NNODE);
  csr_fill<<<gE, b, 0, stream>>>(src, dst, cursor, csrc, ceid);
  reorder_e8<<<gE, b, 0, stream>>>(e1, ceid, e1c);

  // 3 hops, all 8 heads per block; elu on final hop (concat=True)
  hop_l1<<<NNODE, b, 0, stream>>>(H,  H, e1c, rs1, rowptr, csrc, fA, 0);
  hop_l1<<<NNODE, b, 0, stream>>>(fA, H, e1c, rs1, rowptr, csrc, fB, 0);
  hop_l1<<<NNODE, b, 0, stream>>>(fB, H, e1c, rs1, rowptr, csrc, fA, 1); // fA == hcat

  // Layer 2 GEMM: h2[N x 40] = hcat[N x 512] @ W2 (padded to 48 cols)
  wmma_gemm_tdm<1><<<dim3(NNODE / 16), b, 0, stream>>>(
      fA, Bf2, h2, NNODE, HCATW, 3, NCLASS, HCATW, NCLASS);
  node_alpha2<<<gN, b, 0, stream>>>(h2, a2, as2, ad2);
  edge_e1k<<<gE, b, 0, stream>>>(src, dst, as2, ad2, e2, rs2);
  reorder_e1<<<gE, b, 0, stream>>>(e2, ceid, e2c);

  hop_l2<<<(NNODE + 7) / 8, b, 0, stream>>>(h2,  h2, e2c, rs2, rowptr, csrc, f2A);
  hop_l2<<<(NNODE + 7) / 8, b, 0, stream>>>(f2A, h2, e2c, rs2, rowptr, csrc, f2B);
  hop_l2<<<(NNODE + 7) / 8, b, 0, stream>>>(f2B, h2, e2c, rs2, rowptr, csrc, f2A);

  elu_logsoftmax<<<(NNODE + 7) / 8, b, 0, stream>>>(f2A, out);
}
